// ContrastiveNet_59725815218594
// MI455X (gfx1250) — compile-verified, matrix operand order checked by reading, not run on hardware
//
#include <hip/hip_runtime.h>
#include <math.h>

typedef __attribute__((ext_vector_type(16))) __bf16 v16bf;
typedef __attribute__((ext_vector_type(8)))  float  v8f;
typedef __attribute__((ext_vector_type(4)))  unsigned int v4u;
typedef __attribute__((ext_vector_type(8)))  int v8i;
typedef __attribute__((ext_vector_type(4)))  int v4i;

#define B_ROWS 4096
#define D_DIM  2048
#define STRIPE 512
#define NNEG   10
#define TEMPERATURE 0.1f
#define EPSV   1e-8f
#define LDST   40   // ushorts per LDS row = 80B = 64B data + 16B TDM pad, 16B-aligned

union Frag { uint4 q[2]; v16bf v; };

// Split f32 into bf16 hi (RN) + bf16 lo (residual): hi*hi + hi*lo + lo*hi ~ f32 product.
__device__ __forceinline__ void splitbf(float x, unsigned short& hi, unsigned short& lo) {
    unsigned int u = __float_as_uint(x);
    unsigned int r = u + 0x7FFFu + ((u >> 16) & 1u);
    hi = (unsigned short)(r >> 16);
    float hif = __uint_as_float(((unsigned int)hi) << 16);
    lo = (unsigned short)(__float_as_uint(x - hif) >> 16);
}

// ---------------- one-time split of x into bf16 hi/lo planes ----------------
__global__ void split_kernel(const float* __restrict__ x,
                             unsigned short* __restrict__ xhi,
                             unsigned short* __restrict__ xlo) {
    size_t i = ((size_t)blockIdx.x * 256 + threadIdx.x) * 4;
    float4 v = *reinterpret_cast<const float4*>(x + i);
    float f[4] = {v.x, v.y, v.z, v.w};
    ushort4 h, l;
    splitbf(f[0], h.x, l.x); splitbf(f[1], h.y, l.y);
    splitbf(f[2], h.z, l.z); splitbf(f[3], h.w, l.w);
    *reinterpret_cast<ushort4*>(xhi + i) = h;
    *reinterpret_cast<ushort4*>(xlo + i) = l;
}

// ---------------- row norms ----------------
__global__ void norm_kernel(const float* __restrict__ x, float* __restrict__ norms) {
    __shared__ float red[256];
    const int row = blockIdx.x;
    float s = 0.f;
    for (int k = threadIdx.x; k < D_DIM; k += 256) {
        float v = x[(size_t)row * D_DIM + k];
        s += v * v;
    }
    red[threadIdx.x] = s; __syncthreads();
    for (int o = 128; o > 0; o >>= 1) {
        if (threadIdx.x < o) red[threadIdx.x] += red[threadIdx.x + o];
        __syncthreads();
    }
    if (threadIdx.x == 0) norms[row] = sqrtf(red[0]);
}

// ---------------- TDM: 128x32 bf16 tile, global -> padded LDS rows ----------------
// D# per ISA 8.3/8.4: 2-D tile, data_size=2B, pad 4 DWORDs after every 16 DWORDs
// (64B data + 16B pad = 80B LDS row stride, 16B aligned for ds_load_b128 frags).
__device__ __forceinline__ void tdm_load_tile(const unsigned short* src, unsigned lds_addr) {
    unsigned long long ga = (unsigned long long)src;
    v4u g0;
    g0[0] = 1u;                                       // count=1 (valid user D#)
    g0[1] = lds_addr;                                 // lds_addr (bytes)
    g0[2] = (unsigned)(ga & 0xFFFFFFFFu);             // global_addr[31:0]
    g0[3] = (unsigned)((ga >> 32) & 0x1FFFFFFu)       // global_addr[56:32]
          | (2u << 30);                               // type=2 ("image")
    v8i g1;
    g1[0] = (int)((1u << 16)      // data_size = 2 bytes
                | (1u << 20)      // pad_enable
                | (3u << 22)      // pad_interval: 16 DWORDs
                | (3u << 25));    // pad_amount:   4 DWORDs
    g1[1] = (int)(((unsigned)D_DIM & 0xFFFFu) << 16);                 // tensor_dim0[15:0]
    g1[2] = (int)(((unsigned)D_DIM >> 16) |
                  (((unsigned)B_ROWS & 0xFFFFu) << 16));              // dim0[31:16] | dim1[15:0]
    g1[3] = (int)(((unsigned)B_ROWS >> 16) | (32u << 16));            // dim1[31:16] | tile_dim0=32
    g1[4] = (int)128u;                                                // tile_dim1=128, tile_dim2=0
    g1[5] = (int)(unsigned)D_DIM;                                     // tensor_dim0_stride[31:0]
    g1[6] = 0;
    g1[7] = 0;
    v4i z4 = {0, 0, 0, 0};                 // D# groups 2/3: unused for 2-D tile
    v8i z8 = {0, 0, 0, 0, 0, 0, 0, 0};     // extra operand (clang-23 6-arg form)
    __builtin_amdgcn_tensor_load_to_lds(g0, g1, z4, z4, z8, 0);
}

__device__ __forceinline__ unsigned lds_addr_of(const void* p) {
    return (unsigned)(size_t)p;   // generic LDS pointer: addr[31:0] is the LDS offset
}

// ---------------- stripe Gram via split-bf16 WMMA + TDM double buffering ----------------
__global__ __launch_bounds__(512)
void gram_stripe_kernel(const unsigned short* __restrict__ xhi,
                        const unsigned short* __restrict__ xlo,
                        float* __restrict__ gram, int rowBase) {
    __shared__ unsigned short Ahi[2][128 * LDST], Alo[2][128 * LDST];
    __shared__ unsigned short Bhi[2][128 * LDST], Blo[2][128 * LDST];

    const int tid  = threadIdx.x;
    const int lane = tid & 31;
    const int wave = tid >> 5;
    const int waveRow = wave >> 2;          // 0..3
    const int waveCol = wave & 3;           // 0..3
    const int tileRow0 = blockIdx.y * 128;  // row within stripe
    const int tileCol0 = blockIdx.x * 128;  // global column-row index
    const int m16  = lane & 15;
    const int half = lane >> 4;

    const unsigned short* aHi = xhi + (size_t)(rowBase + tileRow0) * D_DIM;
    const unsigned short* aLo = xlo + (size_t)(rowBase + tileRow0) * D_DIM;
    const unsigned short* bHi = xhi + (size_t)tileCol0 * D_DIM;
    const unsigned short* bLo = xlo + (size_t)tileCol0 * D_DIM;

    v8f acc[2][2] = {};

    constexpr int NITER = D_DIM / 32;
    if (wave == 0) {   // wave 0 drives the Tensor Data Mover (EXEC-independent, per-wave)
        tdm_load_tile(aHi, lds_addr_of(&Ahi[0][0]));
        tdm_load_tile(aLo, lds_addr_of(&Alo[0][0]));
        tdm_load_tile(bHi, lds_addr_of(&Bhi[0][0]));
        tdm_load_tile(bLo, lds_addr_of(&Blo[0][0]));
    }

    for (int it = 0; it < NITER; ++it) {
        const int cur = it & 1;
        if (wave == 0) {
            if (it + 1 < NITER) {
                const int nxt = cur ^ 1;
                const int kn = (it + 1) * 32;
                tdm_load_tile(aHi + kn, lds_addr_of(&Ahi[nxt][0]));
                tdm_load_tile(aLo + kn, lds_addr_of(&Alo[nxt][0]));
                tdm_load_tile(bHi + kn, lds_addr_of(&Bhi[nxt][0]));
                tdm_load_tile(bLo + kn, lds_addr_of(&Blo[nxt][0]));
                // TDM ops complete in order per wave: <=4 outstanding means
                // the 4 current-buffer loads have landed in LDS.
                __builtin_amdgcn_s_wait_tensorcnt((short)4);
            } else {
                __builtin_amdgcn_s_wait_tensorcnt((short)0);
            }
        }
        __syncthreads();   // gate all waves on wave 0's tensor-wait

        // Fragment loads per ISA 16-bit A-matrix 16x32 layout:
        // lane -> M=lane%16, K-half=lane/16; VGPR0-3: K=half*8+[0..7],
        // VGPR4-7: K=16+half*8+[0..7] => two 16B LDS reads each.
        Frag ah[2], al[2], bh[2], bl[2];
        #pragma unroll
        for (int i = 0; i < 2; ++i) {
            int ra = (waveRow * 32 + i * 16 + m16) * LDST;
            ah[i].q[0] = *reinterpret_cast<const uint4*>(&Ahi[cur][ra + half * 8]);
            ah[i].q[1] = *reinterpret_cast<const uint4*>(&Ahi[cur][ra + 16 + half * 8]);
            al[i].q[0] = *reinterpret_cast<const uint4*>(&Alo[cur][ra + half * 8]);
            al[i].q[1] = *reinterpret_cast<const uint4*>(&Alo[cur][ra + 16 + half * 8]);
            int rb = (waveCol * 32 + i * 16 + m16) * LDST;
            bh[i].q[0] = *reinterpret_cast<const uint4*>(&Bhi[cur][rb + half * 8]);
            bh[i].q[1] = *reinterpret_cast<const uint4*>(&Bhi[cur][rb + 16 + half * 8]);
            bl[i].q[0] = *reinterpret_cast<const uint4*>(&Blo[cur][rb + half * 8]);
            bl[i].q[1] = *reinterpret_cast<const uint4*>(&Blo[cur][rb + 16 + half * 8]);
        }

        #pragma unroll
        for (int i = 0; i < 2; ++i)
            #pragma unroll
            for (int j = 0; j < 2; ++j) {
                acc[i][j] = __builtin_amdgcn_wmma_f32_16x16x32_bf16(
                    false, ah[i].v, false, bh[j].v, (short)0, acc[i][j], false, false);
                acc[i][j] = __builtin_amdgcn_wmma_f32_16x16x32_bf16(
                    false, ah[i].v, false, bl[j].v, (short)0, acc[i][j], false, false);
                acc[i][j] = __builtin_amdgcn_wmma_f32_16x16x32_bf16(
                    false, al[i].v, false, bh[j].v, (short)0, acc[i][j], false, false);
            }

        __syncthreads();   // all waves done reading `cur` before its next TDM rewrite
    }

    // C/D layout: VGPR r, lanes0-15 -> (M=r, N=lane), lanes16-31 -> (M=r+8, N=lane-16)
    #pragma unroll
    for (int i = 0; i < 2; ++i)
        #pragma unroll
        for (int j = 0; j < 2; ++j) {
            int colg = tileCol0 + waveCol * 32 + j * 16 + m16;
            #pragma unroll
            for (int r = 0; r < 8; ++r) {
                int rowg = tileRow0 + waveRow * 32 + i * 16 + r + half * 8;
                gram[(size_t)rowg * B_ROWS + colg] = acc[i][j][r];
            }
        }
}

// ---------------- per-pair logsumexp gather ----------------
__global__ void gather_kernel(const float* __restrict__ gram, const float* __restrict__ norms,
                              const int* __restrict__ anchor, const int* __restrict__ pos,
                              const int* __restrict__ neg, float* __restrict__ partial,
                              int P, int stripe) {
    int p = blockIdx.x * blockDim.x + threadIdx.x;
    if (p >= P) return;
    int a = anchor[p];
    if ((a / STRIPE) != stripe) return;
    const float* grow = gram + (size_t)(a % STRIPE) * B_ROWS;
    float na = norms[a];
    float l[NNEG + 1];
    #pragma unroll
    for (int j = 0; j <= NNEG; ++j) {
        int c = (j == 0) ? pos[p] : neg[p * NNEG + (j - 1)];
        float dn = fmaxf(na * norms[c], EPSV);
        l[j] = grow[c] / (dn * TEMPERATURE);
    }
    float m = l[0];
    #pragma unroll
    for (int j = 1; j <= NNEG; ++j) m = fmaxf(m, l[j]);
    float se = 0.f;
    #pragma unroll
    for (int j = 0; j <= NNEG; ++j) se += expf(l[j] - m);
    partial[p] = m + logf(se) - l[0];
}

// ---------------- deterministic reduction ----------------
__global__ void reduce1_kernel(const float* __restrict__ partial, float* __restrict__ bsum, int P) {
    __shared__ float red[256];
    float s = 0.f;
    for (int i = blockIdx.x * 256 + threadIdx.x; i < P; i += 128 * 256) s += partial[i];
    red[threadIdx.x] = s; __syncthreads();
    for (int o = 128; o > 0; o >>= 1) {
        if (threadIdx.x < o) red[threadIdx.x] += red[threadIdx.x + o];
        __syncthreads();
    }
    if (threadIdx.x == 0) bsum[blockIdx.x] = red[0];
}

__global__ void reduce2_kernel(const float* __restrict__ bsum, float* __restrict__ out, int P) {
    __shared__ float red[128];
    red[threadIdx.x] = bsum[threadIdx.x]; __syncthreads();
    for (int o = 64; o > 0; o >>= 1) {
        if (threadIdx.x < o) red[threadIdx.x] += red[threadIdx.x + o];
        __syncthreads();
    }
    if (threadIdx.x == 0) out[0] = red[0] / (float)P;
}

extern "C" void kernel_launch(void* const* d_in, const int* in_sizes, int n_in,
                              void* d_out, int out_size, void* d_ws, size_t ws_size,
                              hipStream_t stream) {
    const float* x      = (const float*)d_in[0];
    // d_in[1] = targets (unused: pair indices are precomputed)
    const int*   anchor = (const int*)d_in[2];
    const int*   pos    = (const int*)d_in[3];
    const int*   neg    = (const int*)d_in[4];
    float*       out    = (float*)d_out;
    const int    P      = in_sizes[2];

    const size_t planeBytes = (size_t)B_ROWS * D_DIM * sizeof(unsigned short); // 16 MB
    char*  w       = (char*)d_ws;
    float* norms   = (float*)w;                                     // 16 KB
    unsigned short* xhi = (unsigned short*)(w + 16384);             // 16 MB
    unsigned short* xlo = (unsigned short*)(w + 16384 + planeBytes);
    float* gram    = (float*)(w + 16384 + 2 * planeBytes);          // 8 MB stripe
    float* partial = (float*)(w + 16384 + 2 * planeBytes + (size_t)STRIPE * B_ROWS * 4);
    float* bsum    = partial + P;

    split_kernel<<<(B_ROWS * (D_DIM / 4)) / 256, 256, 0, stream>>>(x, xhi, xlo);
    norm_kernel<<<B_ROWS, 256, 0, stream>>>(x, norms);

    dim3 ggrid(B_ROWS / 128, STRIPE / 128);
    int  gatherBlocks = (P + 255) / 256;
    for (int s = 0; s < B_ROWS / STRIPE; ++s) {
        gram_stripe_kernel<<<ggrid, 512, 0, stream>>>(xhi, xlo, gram, s * STRIPE);
        gather_kernel<<<gatherBlocks, 256, 0, stream>>>(gram, norms, anchor, pos, neg,
                                                        partial, P, s);
    }
    reduce1_kernel<<<128, 256, 0, stream>>>(partial, bsum, P);
    reduce2_kernel<<<1, 128, 0, stream>>>(bsum, out, P);
}
